// GWNet_60713657696867
// MI455X (gfx1250) — matrix operand order repaired and checked by reading, not compiled, not run on hardware
//
#include <hip/hip_runtime.h>
#include <hip/hip_bf16.h>

// ---------------------------------------------------------------------------
// Graph WaveNet forward for MI455X (gfx1250, wave32).
// Dominant cost: GCN diffusion = per-(b,l) (512x512)x(512x32) GEMMs,
// ~335 GFLOP total -> v_wmma_f32_16x16x32_bf16 with f32 accumulation.
// Adjacency pre-transposed to bf16 (L2 resident, 512KB each); H slice staged
// into LDS via global_load_async_to_lds_b128 (ASYNCcnt) then transposed to
// bf16 [c][v] so B fragments are two 16B vector LDS loads.
// Everything else is small 1x1 convs / reductions done with VALU kernels.
// Internal activation layout: channel-last [B][L][N][32].
// ---------------------------------------------------------------------------

#define BB 64
#define NN 512
#define RC 32
#define SC 256
#define EC 512
#define PP 13
#define OUT_LEN 12

typedef __attribute__((ext_vector_type(16))) __bf16 v16bf;
typedef __attribute__((ext_vector_type(8)))  __bf16 v8bf;
typedef __attribute__((ext_vector_type(8)))  float  v8f;

typedef int v4i __attribute__((vector_size(16)));
typedef __attribute__((address_space(1))) v4i* gv4i_p;   // global int4*
typedef __attribute__((address_space(3))) v4i* lv4i_p;   // LDS int4*

#if defined(__gfx1250__) && \
    __has_builtin(__builtin_amdgcn_global_load_async_to_lds_b128) && \
    __has_builtin(__builtin_amdgcn_s_wait_asynccnt)
#define HAS_ASYNC_LDS 1
#else
#define HAS_ASYNC_LDS 0
#endif

__device__ __forceinline__ unsigned short f2bf_bits(float f) {
    union { float f; unsigned u; } v; v.f = f;
    unsigned u = v.u;
    u += 0x7FFFu + ((u >> 16) & 1u);   // round-to-nearest-even
    return (unsigned short)(u >> 16);
}

// ---- start conv: input (B,P,N,2) -> x0 [B][P][N][32] ------------------------
__global__ __launch_bounds__(256) void k_start(const float* __restrict__ in,
                                               const float* __restrict__ sw,
                                               const float* __restrict__ sb,
                                               float* __restrict__ out) {
    int idx = blockIdx.x * 256 + threadIdx.x;          // B*P*N*32, exact grid
    int o = idx & 31;
    int pos = idx >> 5;
    const float* ip = in + pos * 2;
    out[idx] = sw[o * 2 + 0] * ip[0] + sw[o * 2 + 1] * ip[1] + sb[o];
}

// ---- transpose + bf16 convert of a row-normalized adjacency ----------------
__global__ __launch_bounds__(256) void k_adjT(const float* __restrict__ A,
                                              unsigned short* __restrict__ At) {
    int idx = blockIdx.x * 256 + threadIdx.x;          // 512*512, exact grid
    int v = idx >> 9, w = idx & 511;
    At[w * 512 + v] = f2bf_bits(A[v * 512 + w]);
}

// ---- adaptive adjacency: softmax(relu(nv1 @ nv2), axis=1), stored transposed
__global__ __launch_bounds__(256) void k_adp(const float* __restrict__ nv1,
                                             const float* __restrict__ nv2,
                                             unsigned short* __restrict__ At) {
    __shared__ float red[256];
    int v = blockIdx.x;
    int t = threadIdx.x;
    float n1[10];
#pragma unroll
    for (int k = 0; k < 10; ++k) n1[k] = nv1[v * 10 + k];
    float m0 = 0.f, m1 = 0.f;
#pragma unroll
    for (int k = 0; k < 10; ++k) {
        m0 += n1[k] * nv2[k * 512 + t];
        m1 += n1[k] * nv2[k * 512 + t + 256];
    }
    m0 = fmaxf(m0, 0.f);
    m1 = fmaxf(m1, 0.f);
    red[t] = fmaxf(m0, m1);
    __syncthreads();
    for (int s = 128; s > 0; s >>= 1) {
        if (t < s) red[t] = fmaxf(red[t], red[t + s]);
        __syncthreads();
    }
    float mx = red[0];
    __syncthreads();
    float e0 = __expf(m0 - mx), e1 = __expf(m1 - mx);
    red[t] = e0 + e1;
    __syncthreads();
    for (int s = 128; s > 0; s >>= 1) {
        if (t < s) red[t] += red[t + s];
        __syncthreads();
    }
    float inv = 1.f / red[0];
    At[t * 512 + v] = f2bf_bits(e0 * inv);
    At[(t + 256) * 512 + v] = f2bf_bits(e1 * inv);
}

// ---- dilated conv (k=2) + tanh*sigmoid gate --------------------------------
__global__ __launch_bounds__(256) void k_gated(const float* __restrict__ x,
                                               const float* __restrict__ fw,
                                               const float* __restrict__ fb,
                                               const float* __restrict__ gw,
                                               const float* __restrict__ gb,
                                               float* __restrict__ out,
                                               int Lin, int Lout, int d, int total) {
    __shared__ float lfw[2048];   // [o][c][t]
    __shared__ float lgw[2048];
    for (int j = threadIdx.x; j < 2048; j += 256) { lfw[j] = fw[j]; lgw[j] = gw[j]; }
    __syncthreads();
    int idx = blockIdx.x * 256 + threadIdx.x;
    if (idx >= total) return;
    int o = idx & 31;
    int pos = idx >> 5;
    int n = pos & 511;
    int l = (pos >> 9) % Lout;
    int b = pos / (512 * Lout);
    const float* x0 = x + ((size_t)((b * Lin + l) * 512 + n) << 5);
    const float* x1 = x0 + ((size_t)(d * 512) << 5);
    float f = fb[o], g = gb[o];
    const float* wf = lfw + o * 64;
    const float* wg = lgw + o * 64;
#pragma unroll
    for (int c = 0; c < 32; ++c) {
        float a0 = x0[c], a1 = x1[c];
        f += wf[c * 2] * a0 + wf[c * 2 + 1] * a1;
        g += wg[c * 2] * a0 + wg[c * 2 + 1] * a1;
    }
    out[idx] = tanhf(f) * (1.f / (1.f + __expf(-g)));
}

// ---- skip: only the LAST time position of each layer survives to the head --
__global__ __launch_bounds__(256) void k_skip(const float* __restrict__ gated,
                                              const float* __restrict__ sw,
                                              const float* __restrict__ sb,
                                              float* __restrict__ skip,
                                              int Lout, int init) {
    __shared__ float lw[8192];    // 256x32
    for (int j = threadIdx.x; j < 8192; j += 256) lw[j] = sw[j];
    __syncthreads();
    int idx = blockIdx.x * 256 + threadIdx.x;   // B*N*256, exact grid
    int s = idx & 255;
    int bn = idx >> 8;
    int b = bn >> 9, n = bn & 511;
    const float* g = gated + ((size_t)((b * Lout + (Lout - 1)) * 512 + n) << 5);
    float acc = sb[s];
    const float* w = lw + s * 32;
#pragma unroll
    for (int c = 0; c < 32; ++c) acc += w[c] * g[c];
    skip[idx] = init ? acc : (skip[idx] + acc);
}

// ---- GCN per-slice GEMM: Hout[w][c] = sum_v At[w][v] * Hin[v][c]  (WMMA) ---
__global__ __launch_bounds__(256) void k_spmm(const unsigned short* __restrict__ Atbits,
                                              const float* __restrict__ Hin,
                                              float* __restrict__ Hout) {
    __shared__ __attribute__((aligned(16))) float ldsF[512 * 32];          // 64KB fp32 stage
    __shared__ __attribute__((aligned(32))) unsigned short ldsH[32 * 512]; // 32KB bf16 [c][v]
    const int slice = blockIdx.x;
    const float* src = Hin + (size_t)slice * 16384;
    const int tid = threadIdx.x;

    // Stage 1: global -> LDS (fp32).  Async-to-LDS path when available.
#if HAS_ASYNC_LDS
    for (int j = tid * 4; j < 16384; j += 1024) {
        __builtin_amdgcn_global_load_async_to_lds_b128(
            (gv4i_p)(src + j), (lv4i_p)(ldsF + j), 0, 0);
    }
    __builtin_amdgcn_s_wait_asynccnt(0);
#else
    for (int j = tid; j < 16384; j += 256) ldsF[j] = src[j];
#endif
    __syncthreads();

    // Stage 2: transpose + convert -> bf16 [c][v] so B fragments are contiguous.
    for (int j = tid; j < 16384; j += 256) {
        int v = j & 511, c = j >> 9;
        ldsH[c * 512 + v] = f2bf_bits(ldsF[v * 32 + c]);
    }
    __syncthreads();

    const __bf16* lB = reinterpret_cast<const __bf16*>(ldsH);
    const __bf16* gA = reinterpret_cast<const __bf16*>(Atbits);
    const int wave = tid >> 5;
    const int lane = tid & 31;
    const int half = lane >> 4;
    const int lm = lane & 15;

    for (int tile = wave; tile < 64; tile += 8) {    // 32 M-tiles x 2 N-tiles
        const int tN = tile & 1;
        const int tM = tile >> 1;
        const int arow = tM * 16 + lm;               // A-frag row (16-bit layout)
        const int bn = tN * 16 + lm;                 // B-frag column
        const __bf16* gArow = gA + (size_t)arow * 512;
        const __bf16* lBrow = lB + (size_t)bn * 512;
        v8f acc = {};
#pragma unroll 4
        for (int k0 = 0; k0 < 512; k0 += 32) {
            // A 16x32 bf16: lanes 0-15 K={0..7,16..23}, lanes 16-31 K={8..15,24..31}
            v8bf alo = *(const v8bf*)(gArow + k0 + half * 8);
            v8bf ahi = *(const v8bf*)(gArow + k0 + 16 + half * 8);
            v16bf a = __builtin_shufflevector(alo, ahi,
                0, 1, 2, 3, 4, 5, 6, 7, 8, 9, 10, 11, 12, 13, 14, 15);
            // B 32x16 bf16: lanes 0-15 K={0..7,16..23}, lanes 16-31 K={8..15,24..31}
            v8bf blo = *(const v8bf*)(lBrow + k0 + half * 8);
            v8bf bhi = *(const v8bf*)(lBrow + k0 + 16 + half * 8);
            v16bf b = __builtin_shufflevector(blo, bhi,
                0, 1, 2, 3, 4, 5, 6, 7, 8, 9, 10, 11, 12, 13, 14, 15);
            acc = __builtin_amdgcn_wmma_f32_16x16x32_bf16(
                false, a, false, b, (short)0, acc, false, false);
        }
        // C/D: VGPR p -> M = half*8 + p, N = lm
        float* dst = Hout + (size_t)slice * 16384 + (size_t)(tM * 16 + half * 8) * 32 + bn;
#pragma unroll
        for (int p = 0; p < 8; ++p) dst[p * 32] = acc[p];
    }
}

// ---- accumulate y += h * Wk  (chunked 224->32 gcn conv, no concat) ---------
// mode 0: y = bias + dot   mode 1: y = yin + dot   mode 2: y = yin + dot + residual
__global__ __launch_bounds__(256) void k_gcn_acc(const float* __restrict__ h,
                                                 const float* __restrict__ gw,  // stride 224
                                                 const float* __restrict__ gb,
                                                 const float* __restrict__ yin,
                                                 const float* __restrict__ res,
                                                 float* __restrict__ yout,
                                                 int Lout, int Lin, int d,
                                                 int total, int mode) {
    __shared__ float lw[1024];
    for (int j = threadIdx.x; j < 1024; j += 256) lw[j] = gw[(j >> 5) * 224 + (j & 31)];
    __syncthreads();
    int idx = blockIdx.x * 256 + threadIdx.x;
    if (idx >= total) return;
    int o = idx & 31;
    int pos = idx >> 5;
    const float* hp = h + ((size_t)pos << 5);
    float acc = (mode == 0) ? gb[o] : yin[idx];
    const float* w = lw + o * 32;
#pragma unroll
    for (int c = 0; c < 32; ++c) acc += w[c] * hp[c];
    if (mode == 2) {
        int n = pos & 511;
        int l = (pos >> 9) % Lout;
        int b = pos / (512 * Lout);
        acc += res[((size_t)((b * Lin + l + d) * 512 + n) << 5) + o];
    }
    yout[idx] = acc;
}

// ---- BatchNorm (training-mode batch stats) : deterministic 2-stage reduce --
__global__ __launch_bounds__(256) void k_bn_part(const float* __restrict__ x,
                                                 float* __restrict__ part, int NPOS) {
    __shared__ float s1[256];
    __shared__ float s2[256];
    int c = threadIdx.x & 31;
    int g = threadIdx.x >> 5;
    float a = 0.f, b = 0.f;
    for (int p = blockIdx.x * 8 + g; p < NPOS; p += gridDim.x * 8) {
        float v = x[((size_t)p << 5) + c];
        a += v; b += v * v;
    }
    s1[threadIdx.x] = a; s2[threadIdx.x] = b;
    __syncthreads();
    if (threadIdx.x < 32) {
        for (int gg = 1; gg < 8; ++gg) { a += s1[gg * 32 + c]; b += s2[gg * 32 + c]; }
        part[blockIdx.x * 32 + c] = a;
        part[8192 + blockIdx.x * 32 + c] = b;
    }
}

__global__ void k_bn_fin(const float* __restrict__ part,
                         const float* __restrict__ gamma,
                         const float* __restrict__ beta,
                         float* __restrict__ kp, int NPOS) {
    int c = threadIdx.x;
    if (c >= 32) return;
    float s = 0.f, ss = 0.f;
    for (int r = 0; r < 256; ++r) { s += part[r * 32 + c]; ss += part[8192 + r * 32 + c]; }
    float inv = 1.f / (float)NPOS;
    float mean = s * inv;
    float var = ss * inv - mean * mean;
    float istd = rsqrtf(var + 1e-5f);
    float kA = istd * gamma[c];
    kp[c] = kA;
    kp[32 + c] = beta[c] - mean * kA;
}

__global__ __launch_bounds__(256) void k_bn_apply(float* __restrict__ x,
                                                  const float* __restrict__ kp,
                                                  int total) {
    int idx = blockIdx.x * 256 + threadIdx.x;
    if (idx >= total) return;
    int c = idx & 31;
    x[idx] = x[idx] * kp[c] + kp[32 + c];
}

// ---- head: e1 = relu(end1_w @ relu(skip) + b) ------------------------------
__global__ __launch_bounds__(256) void k_end1(const float* __restrict__ skip,
                                              const float* __restrict__ w,
                                              const float* __restrict__ bias,
                                              float* __restrict__ e1) {
    __shared__ float ls[256];
    int idx = blockIdx.x * 256 + threadIdx.x;     // 32768*512, exact grid
    int bn = (blockIdx.x * 256) >> 9;             // whole block shares one (b,n)
    ls[threadIdx.x] = fmaxf(skip[bn * 256 + threadIdx.x], 0.f);
    __syncthreads();
    int o = idx & 511;
    const float* wr = w + (size_t)o * 256;
    float acc = bias[o];
#pragma unroll 4
    for (int c = 0; c < 256; ++c) acc += wr[c] * ls[c];
    e1[idx] = fmaxf(acc, 0.f);
}

__global__ __launch_bounds__(256) void k_end2(const float* __restrict__ e1,
                                              const float* __restrict__ w,
                                              const float* __restrict__ bias,
                                              float* __restrict__ out) {
    int idx = blockIdx.x * 256 + threadIdx.x;     // B*12*N
    if (idx >= BB * OUT_LEN * NN) return;
    int n = idx & 511;
    int t = (idx >> 9) % OUT_LEN;
    int b = idx / (OUT_LEN * NN);
    const float* e = e1 + ((size_t)(b * 512 + n)) * 512;
    const float* wr = w + (size_t)t * 512;
    float acc = bias[t];
#pragma unroll 4
    for (int o = 0; o < 512; ++o) acc += wr[o] * e[o];
    out[idx] = acc;
}

// ---------------------------------------------------------------------------
extern "C" void kernel_launch(void* const* d_in, const int* in_sizes, int n_in,
                              void* d_out, int out_size, void* d_ws, size_t ws_size,
                              hipStream_t stream) {
    (void)in_sizes; (void)n_in; (void)out_size; (void)ws_size;
    const float* input    = (const float*)d_in[0];
    const float* A0       = (const float*)d_in[1];
    const float* A1       = (const float*)d_in[2];
    const float* start_w  = (const float*)d_in[3];
    const float* start_b  = (const float*)d_in[4];
    const float* filter_w = (const float*)d_in[5];
    const float* filter_b = (const float*)d_in[6];
    const float* gate_w   = (const float*)d_in[7];
    const float* gate_b   = (const float*)d_in[8];
    const float* skip_w   = (const float*)d_in[9];
    const float* skip_b   = (const float*)d_in[10];
    const float* gcn_w    = (const float*)d_in[11];
    const float* gcn_b    = (const float*)d_in[12];
    const float* bn_gamma = (const float*)d_in[13];
    const float* bn_beta  = (const float*)d_in[14];
    const float* nodevec1 = (const float*)d_in[15];
    const float* nodevec2 = (const float*)d_in[16];
    const float* end1_w   = (const float*)d_in[17];
    const float* end1_b   = (const float*)d_in[18];
    const float* end2_w   = (const float*)d_in[19];
    const float* end2_b   = (const float*)d_in[20];

    const size_t BUF = (size_t)BB * PP * NN * RC;     // 13,631,488 floats
    float* ws    = (float*)d_ws;
    float* buf0  = ws;
    float* buf1  = ws + BUF;
    float* buf2  = ws + 2 * BUF;
    float* buf3  = ws + 3 * BUF;
    float* skipb = ws + 4 * BUF;                      // B*N*256
    float* bnpart  = skipb + (size_t)BB * NN * SC;    // 2*256*32
    float* bnparam = bnpart + 16384;                  // 64
    unsigned short* adj = (unsigned short*)(bnparam + 64);
    unsigned short* A0t  = adj;
    unsigned short* A1t  = adj + 262144;
    unsigned short* adpt = adj + 524288;

    // prologue
    k_start<<<(BB * PP * NN * RC) / 256, 256, 0, stream>>>(input, start_w, start_b, buf0);
    k_adjT<<<(NN * NN) / 256, 256, 0, stream>>>(A0, A0t);
    k_adjT<<<(NN * NN) / 256, 256, 0, stream>>>(A1, A1t);
    k_adp<<<NN, 256, 0, stream>>>(nodevec1, nodevec2, adpt);

    const int dil[8] = {1, 2, 1, 2, 1, 2, 1, 2};
    int Lin = PP;
    float* xc  = buf0;   // current activation (residual source)
    float* pA  = buf1;   // gated / h ping
    float* pB  = buf2;   // h pong / next x
    float* yac = buf3;   // gcn accumulator

    for (int i = 0; i < 8; ++i) {
        const int d = dil[i];
        const int Lout = Lin - d;
        const int total = BB * Lout * NN * RC;
        const int blocks = (total + 255) / 256;

        k_gated<<<blocks, 256, 0, stream>>>(xc, filter_w + i * 2048, filter_b + i * 32,
                                            gate_w + i * 2048, gate_b + i * 32,
                                            pA, Lin, Lout, d, total);
        k_skip<<<(BB * NN * SC) / 256, 256, 0, stream>>>(pA, skip_w + i * 8192,
                                                         skip_b + i * 256, skipb,
                                                         Lout, i == 0);
        if (i == 7) break;   // last layer's GCN/BN output is dead code

        // chunk 0: y = bias + gated*W0
        k_gcn_acc<<<blocks, 256, 0, stream>>>(pA, gcn_w + i * 7168, gcn_b + i * 32,
                                              nullptr, nullptr, yac,
                                              Lout, Lin, d, total, 0);

        const unsigned short* Aseq[6] = {A0t, A0t, A1t, A1t, adpt, adpt};
        float* hin = pA;
        float* hout = pB;
        float* xnext = nullptr;
        const int S = BB * Lout;
        for (int k = 1; k <= 6; ++k) {
            k_spmm<<<S, 256, 0, stream>>>(Aseq[k - 1], hin, hout);
            if (k == 6) {
                xnext = hin;   // buffer holding h5 is now free -> becomes next x
                k_gcn_acc<<<blocks, 256, 0, stream>>>(hout, gcn_w + i * 7168 + k * 32,
                                                      gcn_b + i * 32, yac, xc, xnext,
                                                      Lout, Lin, d, total, 2);
            } else {
                k_gcn_acc<<<blocks, 256, 0, stream>>>(hout, gcn_w + i * 7168 + k * 32,
                                                      gcn_b + i * 32, yac, nullptr, yac,
                                                      Lout, Lin, d, total, 1);
            }
            float* t = hin; hin = hout; hout = t;
        }

        // batchnorm (batch statistics) on xnext, in place
        const int NPOS = BB * Lout * NN;
        k_bn_part<<<256, 256, 0, stream>>>(xnext, bnpart, NPOS);
        k_bn_fin<<<1, 32, 0, stream>>>(bnpart, bn_gamma + i * 32, bn_beta + i * 32,
                                       bnparam, NPOS);
        k_bn_apply<<<blocks, 256, 0, stream>>>(xnext, bnparam, total);

        // rotate buffers: xnext(==pB) becomes xc; old xc becomes scratch
        float* t = xc; xc = pB; pB = pA; pA = t;
        Lin = Lout;
    }

    // head (all ping-pong buffers dead -> alias e1 onto buf0/buf1 region)
    float* e1 = ws;   // B*N*512 floats = 16.7M, fits in buf0+buf1
    k_end1<<<(BB * NN * EC) / 256, 256, 0, stream>>>(skipb, end1_w, end1_b, e1);
    k_end2<<<(BB * OUT_LEN * NN + 255) / 256, 256, 0, stream>>>(e1, end2_w, end2_b,
                                                                (float*)d_out);
}
